// Mesh_GNN_41936060678384
// MI455X (gfx1250) — matrix operand order repaired and checked by reading, not compiled, not run on hardware
//
#include <hip/hip_runtime.h>
#include <hip/hip_bf16.h>

// ---------------------------------------------------------------------------
// Mesh GNN: edge MLP -> adj @ em_new (dominant, 1 GiB adj) -> node MLP
//   K1: per-wave 16-edge tile, gather+concat rows, f32 WMMA 16x16x4 MLP.
//       Writes em_new (f32, d_out) and emT (bf16 transposed [64][E], d_ws).
//   K2: per-block 16 node rows; 8 waves split K=32768 with bf16 WMMA
//       16x16x32 (adj cvt'd f32->bf16 in-flight); LDS reduction; wave 0
//       runs fused node MLP (f32 WMMA) and stores vm_new.
// Weight staging uses GLOBAL_LOAD_ASYNC_TO_LDS_B128 (ASYNCcnt) — the CDNA5
// direct global->LDS path, no VGPR round trip.
// ---------------------------------------------------------------------------

#define N_NODES 8192
#define N_EDGES 32768
#define LATENT  64
#define HIDDEN  64

typedef float  v2f  __attribute__((ext_vector_type(2)));
typedef float  v4f  __attribute__((ext_vector_type(4)));
typedef float  v8f  __attribute__((ext_vector_type(8)));
typedef __bf16 v16bf __attribute__((ext_vector_type(16)));

// ---- CDNA5 async global->LDS copy (tracked by ASYNCcnt) -------------------
__device__ __forceinline__ void async_g2l_b128(float* lds_dst,
                                               const float* gsrc) {
  unsigned loff = (unsigned)(unsigned long long)lds_dst;  // low 32b = LDS off
  asm volatile("global_load_async_to_lds_b128 %0, %1, off"
               :
               : "v"(loff), "v"(gsrc)
               : "memory");
}

__device__ __forceinline__ void wait_async0() {
#if __has_builtin(__builtin_amdgcn_s_wait_asynccnt)
  __builtin_amdgcn_s_wait_asynccnt(0);
#else
  asm volatile("s_wait_asynccnt 0x0" ::: "memory");
#endif
}

// 4 N-tiles of V_WMMA_F32_16X16X4_F32 sharing one A fragment.
// B layout: lane L -> N = L%16; VGPR0 = W[kr][N], VGPR1 = W[kr+1][N],
// where kr = kk + (L>>4)*2. sW is row-major [K][64] in LDS.
__device__ __forceinline__ void wmma4_f32(v2f a, const float* sW, int kr,
                                          int l16, v8f (&acc)[4]) {
#pragma unroll
  for (int n = 0; n < 4; ++n) {
    v2f b;
    b[0] = sW[kr * 64 + n * 16 + l16];
    b[1] = sW[kr * 64 + 64 + n * 16 + l16];
    acc[n] = __builtin_amdgcn_wmma_f32_16x16x4_f32(
        false, a, false, b, (short)0, acc[n], false, false);
  }
}

// One 64-wide K segment of a 16-row MLP layer. A fragment per lane is a
// contiguous float2 from its row (A 16x4 layout: lanes0-15 K0/K1, 16-31 K2/K3).
__device__ __forceinline__ void mlp_k64(const float* p, const float* sW,
                                        int kbase, int half, int l16,
                                        v8f (&acc)[4]) {
#pragma unroll
  for (int k2 = 0; k2 < 16; ++k2) {
    const int kk = k2 * 4;
    v2f a = *(const v2f*)(p + kk + half * 2);
    wmma4_f32(a, sW, kbase + kk + half * 2, l16, acc);
  }
}

// ===========================================================================
// Kernel 1: edge MLP.  grid = E/128 blocks, 256 threads (8 waves x 16 edges)
// ===========================================================================
__global__ __launch_bounds__(256) void edge_mlp_kernel(
    const float* __restrict__ vm, const float* __restrict__ em,
    const int* __restrict__ esrc, const int* __restrict__ edst,
    const float* __restrict__ W1, const float* __restrict__ b1,
    const float* __restrict__ W2, const float* __restrict__ b2,
    float* __restrict__ em_out, __bf16* __restrict__ emT) {
  extern __shared__ float smem[];
  float* sW1 = smem;                 // 192*64 = 12288 f (48 KiB)
  float* sW2 = sW1 + 192 * 64;       // 64*64  =  4096 f (16 KiB)
  float* sh  = sW2 + 64 * 64;        // 8*16*64 = 8192 f (32 KiB)

  // stage weights straight into LDS via async copies (b128 per lane)
  for (int i = threadIdx.x * 4; i < 192 * 64; i += 256 * 4)
    async_g2l_b128(&sW1[i], &W1[i]);
  for (int i = threadIdx.x * 4; i < 64 * 64; i += 256 * 4)
    async_g2l_b128(&sW2[i], &W2[i]);
  wait_async0();
  __syncthreads();

  const int lane = threadIdx.x & 31;
  const int w    = threadIdx.x >> 5;
  const int half = lane >> 4;   // 0 or 1
  const int l16  = lane & 15;   // row within tile / N within tile
  const int e0   = blockIdx.x * 128 + w * 16;
  const int e    = e0 + l16;

  const int src = esrc[e];
  const int dst = edst[e];
  const float* p_em = em + (size_t)e * LATENT;
  const float* p_s  = vm + (size_t)src * LATENT;
  const float* p_d  = vm + (size_t)dst * LATENT;

  // layer 1: x[192] @ W1[192x64] + b1, accumulators preloaded with bias
  v8f acc[4];
#pragma unroll
  for (int n = 0; n < 4; ++n) {
    const float bb = b1[n * 16 + l16];
#pragma unroll
    for (int j = 0; j < 8; ++j) acc[n][j] = bb;
  }
  mlp_k64(p_em, sW1, 0, half, l16, acc);
  mlp_k64(p_s, sW1, 64, half, l16, acc);
  mlp_k64(p_d, sW1, 128, half, l16, acc);

  // ReLU, stage h (16x64) in per-wave LDS to re-layout C/D -> A fragments
  float* shw = sh + w * (16 * 64);
#pragma unroll
  for (int n = 0; n < 4; ++n)
#pragma unroll
    for (int j = 0; j < 8; ++j) {
      const float hv = fmaxf(acc[n][j], 0.0f);
      shw[(half * 8 + j) * 64 + n * 16 + l16] = hv;
    }

  // layer 2: h[64] @ W2[64x64] + b2
  v8f acc2[4];
#pragma unroll
  for (int n = 0; n < 4; ++n) {
    const float bb = b2[n * 16 + l16];
#pragma unroll
    for (int j = 0; j < 8; ++j) acc2[n][j] = bb;
  }
  mlp_k64(shw + l16 * 64, sW2, 0, half, l16, acc2);

  // store em_new (f32) and transposed bf16 copy emT[c][e]
  float* oe = em_out + (size_t)e0 * HIDDEN;
#pragma unroll
  for (int n = 0; n < 4; ++n)
#pragma unroll
    for (int j = 0; j < 8; ++j) {
      const int r = half * 8 + j;
      const int c = n * 16 + l16;
      const float v = acc2[n][j];
      oe[r * HIDDEN + c] = v;
      emT[(size_t)c * N_EDGES + (e0 + r)] = (__bf16)v;
    }
}

// ===========================================================================
// Kernel 2: edge_sum = adj @ em_new  (bf16 WMMA) fused with node MLP.
// grid = N/16 = 512 blocks, 256 threads (8 waves split K).
// ===========================================================================
__global__ __launch_bounds__(256) void node_kernel(
    const float* __restrict__ adj, const __bf16* __restrict__ emT,
    const float* __restrict__ vm, const float* __restrict__ W1,
    const float* __restrict__ b1, const float* __restrict__ W2,
    const float* __restrict__ b2, float* __restrict__ vm_out) {
  extern __shared__ float smem[];
  float* sW1   = smem;               // 128*64 = 8192 f (32 KiB)
  float* sW2   = sW1 + 128 * 64;     // 64*64  = 4096 f (16 KiB)
  float* spart = sW2 + 64 * 64;      // 8*16*64 = 8192 f (32 KiB)
  float* ssum  = spart + 8 * 16 * 64;  // 16*64 = 1024 f (4 KiB)
  float* sh    = ssum + 16 * 64;       // 16*64 = 1024 f (4 KiB)

  for (int i = threadIdx.x * 4; i < 128 * 64; i += 256 * 4)
    async_g2l_b128(&sW1[i], &W1[i]);
  for (int i = threadIdx.x * 4; i < 64 * 64; i += 256 * 4)
    async_g2l_b128(&sW2[i], &W2[i]);
  wait_async0();
  __syncthreads();

  const int lane = threadIdx.x & 31;
  const int w    = threadIdx.x >> 5;
  const int half = lane >> 4;
  const int l16  = lane & 15;
  const int m0   = blockIdx.x * 16;

  // ---- adj[m0:m0+16, :] @ em_new -> 16x64, K split across 8 waves -------
  const float* arow = adj + (size_t)(m0 + l16) * N_EDGES;
  v8f acc[4];
#pragma unroll
  for (int n = 0; n < 4; ++n)
#pragma unroll
    for (int j = 0; j < 8; ++j) acc[n][j] = 0.0f;

#pragma unroll 2
  for (int kb = w; kb < N_EDGES / 32; kb += 8) {
    const int k0 = kb * 32;
    // A (16x32 bf16) layout: lane holds K = base..base+15,
    // base = k0 + half*16.  Source: 16 f32 -> cvt to bf16.
    const float* ap = arow + k0 + half * 16;
    v4f f0 = *(const v4f*)(ap);
    v4f f1 = *(const v4f*)(ap + 4);
    v4f f2 = *(const v4f*)(ap + 8);
    v4f f3 = *(const v4f*)(ap + 12);
    __builtin_prefetch(ap + 256, 0, 0);  // next K chunk for this wave
    v16bf a;
#pragma unroll
    for (int i = 0; i < 4; ++i) {
      a[i]      = (__bf16)f0[i];
      a[4 + i]  = (__bf16)f1[i];
      a[8 + i]  = (__bf16)f2[i];
      a[12 + i] = (__bf16)f3[i];
    }
    // B (32x16 bf16): lane L -> N = L%16, K = k0 + (L>>4)*16 .. +15,
    // contiguous in emT (transposed em_new).
#pragma unroll
    for (int n = 0; n < 4; ++n) {
      const __bf16* bp =
          emT + (size_t)(n * 16 + l16) * N_EDGES + k0 + half * 16;
      v16bf b = *(const v16bf*)bp;
      acc[n] = __builtin_amdgcn_wmma_f32_16x16x32_bf16(
          false, a, false, b, (short)0, acc[n], false, false);
    }
  }

  // ---- cross-wave reduction of the 8 partial 16x64 tiles via LDS --------
#pragma unroll
  for (int n = 0; n < 4; ++n)
#pragma unroll
    for (int j = 0; j < 8; ++j)
      spart[w * 1024 + (half * 8 + j) * 64 + n * 16 + l16] = acc[n][j];
  __syncthreads();
  for (int i = threadIdx.x; i < 1024; i += 256) {
    float s = 0.0f;
#pragma unroll
    for (int w2 = 0; w2 < 8; ++w2) s += spart[w2 * 1024 + i];
    ssum[i] = s;
  }
  __syncthreads();

  // ---- fused node MLP on rows m0..m0+15 (wave 0, uniform EXEC) ----------
  if (w == 0) {
    v8f a1[4];
#pragma unroll
    for (int n = 0; n < 4; ++n) {
      const float bb = b1[n * 16 + l16];
#pragma unroll
      for (int j = 0; j < 8; ++j) a1[n][j] = bb;
    }
    // K 0..63: vm_updated rows (global);  K 64..127: edge_sum (LDS)
    mlp_k64(vm + (size_t)(m0 + l16) * LATENT, sW1, 0, half, l16, a1);
    mlp_k64(ssum + l16 * 64, sW1, 64, half, l16, a1);

    // ReLU -> LDS re-layout
#pragma unroll
    for (int n = 0; n < 4; ++n)
#pragma unroll
      for (int j = 0; j < 8; ++j)
        sh[(half * 8 + j) * 64 + n * 16 + l16] = fmaxf(a1[n][j], 0.0f);

    v8f a2[4];
#pragma unroll
    for (int n = 0; n < 4; ++n) {
      const float bb = b2[n * 16 + l16];
#pragma unroll
      for (int j = 0; j < 8; ++j) a2[n][j] = bb;
    }
    mlp_k64(sh + l16 * 64, sW2, 0, half, l16, a2);

#pragma unroll
    for (int n = 0; n < 4; ++n)
#pragma unroll
      for (int j = 0; j < 8; ++j)
        vm_out[(size_t)(m0 + half * 8 + j) * HIDDEN + n * 16 + l16] =
            a2[n][j];
  }
}

// ===========================================================================
extern "C" void kernel_launch(void* const* d_in, const int* in_sizes, int n_in,
                              void* d_out, int out_size, void* d_ws,
                              size_t ws_size, hipStream_t stream) {
  const float* vm   = (const float*)d_in[0];   // [N, 64]
  const float* em   = (const float*)d_in[1];   // [E, 64]
  const float* adj  = (const float*)d_in[2];   // [N, E]
  const int*   esrc = (const int*)d_in[3];     // [E]
  const int*   edst = (const int*)d_in[4];     // [E]
  const float* emW1 = (const float*)d_in[5];   // [192, 64]
  const float* emb1 = (const float*)d_in[6];
  const float* emW2 = (const float*)d_in[7];   // [64, 64]
  const float* emb2 = (const float*)d_in[8];
  const float* vmW1 = (const float*)d_in[9];   // [128, 64]
  const float* vmb1 = (const float*)d_in[10];
  const float* vmW2 = (const float*)d_in[11];  // [64, 64]
  const float* vmb2 = (const float*)d_in[12];

  float* out_vm = (float*)d_out;                         // [N, 64]
  float* out_em = out_vm + (size_t)N_NODES * HIDDEN;     // [E, 64]
  __bf16* emT   = (__bf16*)d_ws;                         // [64, E] bf16

  const size_t shA = (192 * 64 + 64 * 64 + 8 * 16 * 64) * sizeof(float);
  edge_mlp_kernel<<<N_EDGES / 128, 256, shA, stream>>>(
      vm, em, esrc, edst, emW1, emb1, emW2, emb2, out_em, emT);

  const size_t shB =
      (128 * 64 + 64 * 64 + 8 * 16 * 64 + 16 * 64 + 16 * 64) * sizeof(float);
  node_kernel<<<N_NODES / 16, 256, shB, stream>>>(
      adj, emT, vm, vmW1, vmb1, vmW2, vmb2, out_vm);
}
